// SkipGramNS_17523466568402
// MI455X (gfx1250) — compile-verified, matrix operand order checked by reading, not run on hardware
//
#include <hip/hip_runtime.h>

// Skip-gram negative-sampling loss on MI455X (gfx1250, wave32).
//
// Roofline: ~88 MFLOP vs ~180 MB of randomly gathered 512B embedding rows
// -> latency/gather bound (~8us floor at 23.3 TB/s; tables fit in 192MB L2).
// Strategy: 16 batch rows per tile; dot products ride the WMMA pipe
// (v_wmma_f32_16x16x32_f16 chained over K=128, diagonal extraction) so the
// VALU only does address math + log-sigmoid. Two waves split each tile's 21
// scores for 2x memory-level parallelism; the B-tile gather is software-
// pipelined one score ahead (unroll-2 ping-pong, no copy movs); per-wave
// scalars are forced uniform via readfirstlane so control flow stays on the
// SALU; wave-level shfl_xor reduction leaves one global_atomic_add_f32/wave.

typedef __attribute__((ext_vector_type(16))) _Float16 v16h;
typedef __attribute__((ext_vector_type(8)))  float    v8f;

#define DIM    128
#define BATCH  16384
#define NEG    20
#define TILES  (BATCH / 16)
#define SPLIT  2                 // waves cooperating per tile (score split 11+10)
#define UNITS  (TILES * SPLIT)

// One K=32 chunk of a 128-float row in 16-bit WMMA operand layout:
// lanes 0-15 hold K {k0+0..7, k0+16..23}; lanes 16-31 hold K {k0+8..15, k0+24..31}.
__device__ __forceinline__ v16h load_chunk(const float* __restrict__ row, int k0, int hi) {
    const int base = k0 + (hi ? 8 : 0);
    const float4* q0 = (const float4*)(row + base);
    const float4* q1 = (const float4*)(row + base + 16);
    float4 f0 = q0[0];
    float4 f1 = q0[1];
    float4 f2 = q1[0];
    float4 f3 = q1[1];
    v16h h;
    h[0]  = (_Float16)f0.x; h[1]  = (_Float16)f0.y; h[2]  = (_Float16)f0.z; h[3]  = (_Float16)f0.w;
    h[4]  = (_Float16)f1.x; h[5]  = (_Float16)f1.y; h[6]  = (_Float16)f1.z; h[7]  = (_Float16)f1.w;
    h[8]  = (_Float16)f2.x; h[9]  = (_Float16)f2.y; h[10] = (_Float16)f2.z; h[11] = (_Float16)f2.w;
    h[12] = (_Float16)f3.x; h[13] = (_Float16)f3.y; h[14] = (_Float16)f3.z; h[15] = (_Float16)f3.w;
    return h;
}

struct BTile { v16h b0, b1, b2, b3; };

__device__ __forceinline__ BTile load_tile(const float* __restrict__ row, int hi) {
    BTile t;
    t.b0 = load_chunk(row,  0, hi);
    t.b1 = load_chunk(row, 32, hi);
    t.b2 = load_chunk(row, 64, hi);
    t.b3 = load_chunk(row, 96, hi);
    return t;
}

// Branch-free 8-way select of one accumulator element (EXEC stays uniform).
__device__ __forceinline__ float diag_extract(v8f c, int sel) {
    float r =               (float)c[0];
    r = (sel == 1) ? (float)c[1] : r;
    r = (sel == 2) ? (float)c[2] : r;
    r = (sel == 3) ? (float)c[3] : r;
    r = (sel == 4) ? (float)c[4] : r;
    r = (sel == 5) ? (float)c[5] : r;
    r = (sel == 6) ? (float)c[6] : r;
    r = (sel == 7) ? (float)c[7] : r;
    return r;
}

__device__ __forceinline__ float log_sigmoid(float x) {
    // x clipped to [-10, 10]; exp(10) ~ 2.2e4, safe in f32.
    return -__logf(1.0f + __expf(-x));
}

__global__ void __launch_bounds__(128) sgns_zero(float* out) {
    if (threadIdx.x == 0 && blockIdx.x == 0) out[0] = 0.0f;
}

__global__ void __launch_bounds__(128) sgns_wmma_kernel(
    const int*   __restrict__ input_pos,
    const int*   __restrict__ output_pos,
    const int*   __restrict__ output_neg,
    const float* __restrict__ in_emb,
    const float* __restrict__ out_emb,
    float*       __restrict__ out)
{
    const int lane  = threadIdx.x & 31;
    const int wave  = threadIdx.x >> 5;
    const int wpb   = blockDim.x >> 5;
    const int gwave = blockIdx.x * wpb + wave;
    const int nwave = gridDim.x * wpb;

    const int  col   = lane & 15;   // tile row this lane serves (A: M, B: N)
    const int  hi    = lane >> 4;   // K half selector within each 32-wide chunk
    const int  sel   = lane & 7;
    // Diagonal D[i][i]: rows 0-7 -> lanes 0-7 (acc[lane]); rows 8-15 -> lanes 24-31 (acc[lane-24]).
    const bool valid = (lane < 8) || (lane >= 24);

    float lane_sum = 0.0f;

    #pragma unroll 1
    for (int u = gwave; u < UNITS; u += nwave) {
        // Per-wave-uniform scalars: force into SGPRs so control flow and index
        // base math run on the SALU (no EXEC juggling around the WMMAs).
        const int unit   = __builtin_amdgcn_readfirstlane(u);
        const int tile   = unit / SPLIT;
        const int half   = unit % SPLIT;
        const int rbase  = tile * 16;
        const int nscore = (half == 0) ? (1 + NEG / 2) : (NEG / 2);
        const float sign0 = (half == 0) ? 1.0f : -1.0f;   // slot 0 is pos only in half 0

        // y-index for local score slot s of this wave's share.
        // half 0: s=0 -> pos, s=1..10 -> neg 0..9 ; half 1: s=0..9 -> neg 10..19.
        auto yindex = [&](int s) -> int {
            if (half == 0) {
                return (s == 0) ? output_pos[rbase + col]
                                : output_neg[(size_t)(rbase + col) * NEG + (s - 1)];
            }
            return output_neg[(size_t)(rbase + col) * NEG + (10 + s)];
        };

        // A tile: 16 gathered input vectors, held live across the score loop.
        const int xi = input_pos[rbase + col];
        const float* xrow = in_emb + (size_t)xi * DIM;
        const v16h A0 = load_chunk(xrow,  0, hi);
        const v16h A1 = load_chunk(xrow, 32, hi);
        const v16h A2 = load_chunk(xrow, 64, hi);
        const v16h A3 = load_chunk(xrow, 96, hi);

        // Software pipeline: gather B tile for s+1 while WMMA/epilogue run on s.
        // unroll 2 -> register ping-pong instead of cur=nxt copy movs.
        BTile cur = load_tile(out_emb + (size_t)yindex(0) * DIM, hi);

        #pragma unroll 2
        for (int s = 0; s < nscore; ++s) {
            BTile nxt = cur;
            if (s + 1 < nscore) {                 // scalar branch (s, nscore uniform)
                nxt = load_tile(out_emb + (size_t)yindex(s + 1) * DIM, hi);
            }

            v8f c = {};
            c = __builtin_amdgcn_wmma_f32_16x16x32_f16(false, A0, false, cur.b0, (short)0, c, false, false);
            c = __builtin_amdgcn_wmma_f32_16x16x32_f16(false, A1, false, cur.b1, (short)0, c, false, false);
            c = __builtin_amdgcn_wmma_f32_16x16x32_f16(false, A2, false, cur.b2, (short)0, c, false, false);
            c = __builtin_amdgcn_wmma_f32_16x16x32_f16(false, A3, false, cur.b3, (short)0, c, false, false);

            float score = diag_extract(c, sel);
            score = fminf(10.0f, fmaxf(-10.0f, score));
            const float sgn = (s == 0) ? sign0 : -1.0f;   // uniform select
            const float ls  = log_sigmoid(sgn * score);
            lane_sum += valid ? ls : 0.0f;

            cur = nxt;
        }
    }

    // Wave-level reduction (invalid lanes hold 0) -> one atomic per wave.
    #pragma unroll
    for (int off = 16; off >= 1; off >>= 1) {
        lane_sum += __shfl_xor(lane_sum, off, 32);
    }
    if (lane == 0) {
        atomicAdd(out, -lane_sum * (1.0f / (float)BATCH));
    }
}

extern "C" void kernel_launch(void* const* d_in, const int* in_sizes, int n_in,
                              void* d_out, int out_size, void* d_ws, size_t ws_size,
                              hipStream_t stream) {
    (void)in_sizes; (void)n_in; (void)out_size; (void)d_ws; (void)ws_size;
    const int*   input_pos  = (const int*)  d_in[0];
    const int*   output_pos = (const int*)  d_in[1];
    const int*   output_neg = (const int*)  d_in[2];
    const float* in_emb     = (const float*)d_in[3];
    const float* out_emb    = (const float*)d_in[4];
    float*       out        = (float*)d_out;

    sgns_zero<<<1, 128, 0, stream>>>(out);

    // 512 blocks x 4 waves = 2048 waves -> one (tile, score-half) unit each.
    sgns_wmma_kernel<<<512, 128, 0, stream>>>(input_pos, output_pos, output_neg,
                                              in_emb, out_emb, out);
}
